// LSTM_31404800868365
// MI455X (gfx1250) — matrix operand order repaired and checked by reading, not compiled, not run on hardware
//
#include <hip/hip_runtime.h>
#include <hip/hip_bf16.h>

// ---------------- types ----------------
typedef __attribute__((ext_vector_type(16))) __bf16 v16bf;
typedef __attribute__((ext_vector_type(8)))  __bf16 v8bf;
typedef __attribute__((ext_vector_type(8)))  float  v8f;
typedef __attribute__((address_space(3))) const char* lds_cptr;

// ---------------- problem constants ----------------
#define Bsz 128
#define Ssz 1024
#define Isz 512
#define Hsz 1024
#define G3  3072          // 3*H gate columns
#define K0  1536          // I + H   (layer 0 reduction)
#define K1  2048          // 2 * H   (layer 1 reduction)

// ---------------- tiling ----------------
#define MBLK     16       // batch rows per workgroup (one WMMA M tile)
#define SLICE    128      // hidden units per workgroup (16 per wave)
#define NMB      (Bsz / MBLK)    // 8
#define NSL      (Hsz / SLICE)   // 8
#define NWG      (NMB * NSL)     // 64 persistent workgroups
#define NTHREADS 256             // 8 waves (wave32)

__device__ __forceinline__ float fsigmoid(float x) {
    return 1.0f / (1.0f + __expf(-x));
}

// LDS byte offset of a generic pointer into shared memory (addrspacecast).
__device__ __forceinline__ unsigned lds_off(const void* p) {
    return (unsigned)(unsigned long long)(lds_cptr)p;
}

// CDNA5 async DMA: 16 bytes global -> LDS, tracked by ASYNCcnt, no VGPR data.
// Device scope: h-state ping-pong buffers are written by other WGPs.
__device__ __forceinline__ void async_b128(unsigned ldsOff, const void* g) {
    asm volatile("global_load_async_to_lds_b128 %0, %1, off scope:SCOPE_DEV"
                 :: "v"(ldsOff), "v"((unsigned long long)g)
                 : "memory");
}
__device__ __forceinline__ void wait_async() {
    asm volatile("s_wait_asynccnt 0x0" ::: "memory");
}

// Device-wide barrier for the persistent grid. Counter never resets within a
// launch; barrier k waits until count >= (k+1)*NWG. Reset by prep kernel.
__device__ __forceinline__ void grid_sync(unsigned* bar, unsigned target) {
    __syncthreads();
    if (threadIdx.x == 0) {
        __threadfence();                 // release: h-state visible device-wide
        atomicAdd(bar, 1u);
        while (atomicAdd(bar, 0u) < target) {
            __builtin_amdgcn_s_sleep(8);
        }
    }
    __syncthreads();
    __threadfence();                     // acquire: invalidate stale WGP$ lines
}

// ---------------------------------------------------------------------------
// Prep: fp32 weights -> bf16 (L2-resident working set), zero t=0 hidden
// buffers, reset the grid barrier counter. Runs once per kernel_launch.
// ---------------------------------------------------------------------------
__global__ void lstm_prep(const float* __restrict__ W0, const float* __restrict__ W1,
                          __bf16* __restrict__ Wb0, __bf16* __restrict__ Wb1,
                          __bf16* __restrict__ h0buf, __bf16* __restrict__ h1buf,
                          unsigned* __restrict__ bar) {
    const long n0 = (long)G3 * K0;
    const long n1 = (long)G3 * K1;
    const long nh = (long)Bsz * Hsz;
    const long stride = (long)gridDim.x * blockDim.x;
    const long start  = (long)blockIdx.x * blockDim.x + threadIdx.x;
    for (long i = start; i < n1; i += stride) {
        Wb1[i] = (__bf16)W1[i];
        if (i < n0) Wb0[i] = (__bf16)W0[i];
    }
    for (long i = start; i < nh; i += stride) {   // parity-0 buffers read at t=0
        h0buf[i] = (__bf16)0.0f;
        h1buf[i] = (__bf16)0.0f;
    }
    if (start == 0) *bar = 0u;
}

// ---------------------------------------------------------------------------
// Persistent 2-layer LSTM. One grid barrier per timestep; ping-pong h buffers
// remove all other cross-step hazards. c0/c1 cell state lives in registers.
// Staging is software-pipelined: h1_prev DMA overlaps the layer-0 cell update
// and is drained before the barrier; x_{t+1} staging overlaps the layer-1 cell update.
// ---------------------------------------------------------------------------
__global__ void __launch_bounds__(NTHREADS, 1)
lstm_persistent(const float* __restrict__ x,   const float* __restrict__ fw,
                const float* __restrict__ tw,  const float* __restrict__ mag,
                const float* __restrict__ b0,  const float* __restrict__ b1,
                const __bf16* __restrict__ Wb0, const __bf16* __restrict__ Wb1,
                __bf16* __restrict__ h0buf, __bf16* __restrict__ h1buf,
                float* __restrict__ out, unsigned* __restrict__ bar) {
    __shared__ __bf16 Abuf[MBLK * K1];          // 64 KB activation staging

    const int tid  = threadIdx.x;
    const int wave = tid >> 5;
    const int lane = tid & 31;
    const int ln   = lane & 15;                 // N / M sub-index
    const int lh   = lane >> 4;                 // half-wave select
    const int mb   = blockIdx.x & (NMB - 1);
    const int sl   = blockIdx.x >> 3;
    const int bbase = mb * MBLK;
    const int hbase = sl * SLICE + wave * 16;   // this wave's 16 hidden columns
    const int cI = hbase, cG = Hsz + hbase, cO = 2 * Hsz + hbase;

    const float emag = __expf(mag[0]);
    const float bI0 = b0[cI + ln], bG0 = b0[cG + ln], bO0 = b0[cO + ln];
    const float bI1 = b1[cI + ln], bG1 = b1[cG + ln], bO1 = b1[cO + ln];

    const v8f zf = {0.f, 0.f, 0.f, 0.f, 0.f, 0.f, 0.f, 0.f};
    v8f c0 = zf, c1 = zf;                       // register-resident cell state

    // A-fragment base: lane holds row M=ln; K chunks {k+lh*16, k+lh*16+8}
    const __bf16* Arow = &Abuf[ln * K1];

    // ---- prologue: stage scaled x(0) into cols [0, Isz) ----
    {
        const float sct = tw[0] * emag;
        for (int idx = tid; idx < MBLK * Isz; idx += NTHREADS) {
            const int m = idx >> 9, i = idx & (Isz - 1);
            Abuf[m * K1 + i] =
                (__bf16)(x[(size_t)(bbase + m) * Ssz * Isz + i] * fw[i] * sct);
        }
    }

    for (int t = 0; t < Ssz; ++t) {
        const int rp = t & 1, wp = (t + 1) & 1;
        const __bf16* h0r = h0buf + (size_t)rp * Bsz * Hsz;
        __bf16*       h0w = h0buf + (size_t)wp * Bsz * Hsz;
        const __bf16* h1r = h1buf + (size_t)rp * Bsz * Hsz;
        __bf16*       h1w = h1buf + (size_t)wp * Bsz * Hsz;

        // ---- async DMA h0_prev into cols [Isz, K0) ----
        for (int c = tid; c < (MBLK * Hsz) / 8; c += NTHREADS) {
            const int m = c >> 7, j = (c & 127) * 8;   // 16B chunks
            async_b128(lds_off(&Abuf[m * K1 + Isz + j]),
                       h0r + (size_t)(bbase + m) * Hsz + j);
        }
        wait_async();
        __syncthreads();

        // ---- layer-0 GEMM: gates[16 x 48] per wave (i,g,o tiles) ----
        v8f aI = zf, aG = zf, aO = zf;
        {
            const __bf16* BI = Wb0 + (size_t)(cI + ln) * K0 + lh * 16;
            const __bf16* BG = Wb0 + (size_t)(cG + ln) * K0 + lh * 16;
            const __bf16* BO = Wb0 + (size_t)(cO + ln) * K0 + lh * 16;
            for (int k = 0; k < K0; k += 32) {
                const v8bf a0 = *(const v8bf*)(Arow + k + lh * 16);
                const v8bf a1 = *(const v8bf*)(Arow + k + 8 + lh * 16);
                const v16bf a = __builtin_shufflevector(
                    a0, a1, 0, 1, 2, 3, 4, 5, 6, 7, 8, 9, 10, 11, 12, 13, 14, 15);
                const v16bf wi = *(const v16bf*)(BI + k);
                const v16bf wg = *(const v16bf*)(BG + k);
                const v16bf wo = *(const v16bf*)(BO + k);
                __builtin_prefetch(BI + k + 256, 0, 1);   // global_prefetch_b8
                __builtin_prefetch(BG + k + 256, 0, 1);
                __builtin_prefetch(BO + k + 256, 0, 1);
                aI = __builtin_amdgcn_wmma_f32_16x16x32_bf16(
                    false, a, false, wi, (short)0, aI, false, false);
                aG = __builtin_amdgcn_wmma_f32_16x16x32_bf16(
                    false, a, false, wg, (short)0, aG, false, false);
                aO = __builtin_amdgcn_wmma_f32_16x16x32_bf16(
                    false, a, false, wo, (short)0, aO, false, false);
            }
        }
        __syncthreads();     // all waves done reading Abuf

        // ---- prestage h1_prev into cols [Hsz, 2*Hsz): DMA flies during the
        //      cell update below; drained before the grid barrier ----
        for (int c = tid; c < (MBLK * Hsz) / 8; c += NTHREADS) {
            const int m = c >> 7, j = (c & 127) * 8;
            async_b128(lds_off(&Abuf[m * K1 + Hsz + j]),
                       h1r + (size_t)(bbase + m) * Hsz + j);
        }

        // ---- layer-0 cell update (register-level; c0 never leaves VGPRs) ----
#pragma unroll
        for (int r = 0; r < 8; ++r) {
            const float iv = aI[r] + bI0;
            const float gv = aG[r] + bG0;
            const float ov = aO[r] + bO0;
            const float cp = c0[r];
            const float f  = fsigmoid(cp);                 // module quirk
            const float cn = f * cp + fsigmoid(iv) * tanhf(gv);
            const float hn = fsigmoid(ov) * tanhf(cn) * f; // module quirk
            c0[r] = cn;
            const int m = r + 8 * lh;
            h0w[(size_t)(bbase + m) * Hsz + hbase + ln] = (__bf16)hn;
            if (t == Ssz - 1) {
                const size_t fb = (size_t)Bsz * Ssz * Hsz;
                const size_t o  = (size_t)(bbase + m) * Hsz + hbase + ln;
                out[fb + o]                      = hn;     // final h0
                out[fb + (size_t)Bsz * Hsz + o]  = cn;     // final c0
            }
        }
        wait_async();

        // ---- device-wide barrier: h0(t) visible to every hidden slice ----
        grid_sync(bar, (unsigned)(t + 1) * NWG);

        // ---- async DMA h0_new into cols [0, Hsz) ----
        for (int c = tid; c < (MBLK * Hsz) / 8; c += NTHREADS) {
            const int m = c >> 7, j = (c & 127) * 8;
            async_b128(lds_off(&Abuf[m * K1 + j]),
                       h0w + (size_t)(bbase + m) * Hsz + j);
        }
        wait_async();
        __syncthreads();

        // ---- layer-1 GEMM ----
        aI = zf; aG = zf; aO = zf;
        {
            const __bf16* BI = Wb1 + (size_t)(cI + ln) * K1 + lh * 16;
            const __bf16* BG = Wb1 + (size_t)(cG + ln) * K1 + lh * 16;
            const __bf16* BO = Wb1 + (size_t)(cO + ln) * K1 + lh * 16;
            for (int k = 0; k < K1; k += 32) {
                const v8bf a0 = *(const v8bf*)(Arow + k + lh * 16);
                const v8bf a1 = *(const v8bf*)(Arow + k + 8 + lh * 16);
                const v16bf a = __builtin_shufflevector(
                    a0, a1, 0, 1, 2, 3, 4, 5, 6, 7, 8, 9, 10, 11, 12, 13, 14, 15);
                const v16bf wi = *(const v16bf*)(BI + k);
                const v16bf wg = *(const v16bf*)(BG + k);
                const v16bf wo = *(const v16bf*)(BO + k);
                __builtin_prefetch(BI + k + 256, 0, 1);
                __builtin_prefetch(BG + k + 256, 0, 1);
                __builtin_prefetch(BO + k + 256, 0, 1);
                aI = __builtin_amdgcn_wmma_f32_16x16x32_bf16(
                    false, a, false, wi, (short)0, aI, false, false);
                aG = __builtin_amdgcn_wmma_f32_16x16x32_bf16(
                    false, a, false, wg, (short)0, aG, false, false);
                aO = __builtin_amdgcn_wmma_f32_16x16x32_bf16(
                    false, a, false, wo, (short)0, aO, false, false);
            }
        }
        __syncthreads();     // all waves done reading Abuf

        // ---- layer-1 cell update + outputs ----
#pragma unroll
        for (int r = 0; r < 8; ++r) {
            const float iv = aI[r] + bI1;
            const float gv = aG[r] + bG1;
            const float ov = aO[r] + bO1;
            const float cp = c1[r];
            const float f  = fsigmoid(cp);
            const float cn = f * cp + fsigmoid(iv) * tanhf(gv);
            const float hn = fsigmoid(ov) * tanhf(cn) * f;
            c1[r] = cn;
            const int m = r + 8 * lh;
            h1w[(size_t)(bbase + m) * Hsz + hbase + ln] = (__bf16)hn;
            out[((size_t)(bbase + m) * Ssz + t) * Hsz + hbase + ln] = hn;
            if (t == Ssz - 1) {
                const size_t fb = (size_t)Bsz * Ssz * Hsz + 2 * (size_t)Bsz * Hsz;
                const size_t o  = (size_t)(bbase + m) * Hsz + hbase + ln;
                out[fb + o]                      = hn;     // final h1
                out[fb + (size_t)Bsz * Hsz + o]  = cn;     // final c1
            }
        }

        // ---- prestage scaled x(t+1) into cols [0, Isz): overlaps the writes
        //      above; consumed after next iteration's staging barrier ----
        if (t + 1 < Ssz) {
            const float sct = tw[t + 1] * emag;
            for (int idx = tid; idx < MBLK * Isz; idx += NTHREADS) {
                const int m = idx >> 9, i = idx & (Isz - 1);
                Abuf[m * K1 + i] = (__bf16)(
                    x[((size_t)(bbase + m) * Ssz + (t + 1)) * Isz + i] * fw[i] * sct);
            }
        }
    }
}

// ---------------------------------------------------------------------------
extern "C" void kernel_launch(void* const* d_in, const int* in_sizes, int n_in,
                              void* d_out, int out_size, void* d_ws, size_t ws_size,
                              hipStream_t stream) {
    const float* x  = (const float*)d_in[0];
    const float* fw = (const float*)d_in[1];
    const float* tw = (const float*)d_in[2];
    const float* mg = (const float*)d_in[3];
    const float* W0 = (const float*)d_in[4];
    const float* b0 = (const float*)d_in[5];
    const float* W1 = (const float*)d_in[6];
    const float* b1 = (const float*)d_in[7];

    char* ws = (char*)d_ws;
    size_t off = 0;
    auto take = [&](size_t bytes) -> char* {
        char* p = ws + off;
        off = (off + bytes + 255) & ~(size_t)255;
        return p;
    };
    __bf16* Wb0  = (__bf16*)take((size_t)G3 * K0 * sizeof(__bf16));
    __bf16* Wb1  = (__bf16*)take((size_t)G3 * K1 * sizeof(__bf16));
    __bf16* h0b  = (__bf16*)take((size_t)2 * Bsz * Hsz * sizeof(__bf16));
    __bf16* h1b  = (__bf16*)take((size_t)2 * Bsz * Hsz * sizeof(__bf16));
    unsigned* bar = (unsigned*)take(256);

    lstm_prep<<<512, 256, 0, stream>>>(W0, W1, Wb0, Wb1, h0b, h1b, bar);
    lstm_persistent<<<NWG, NTHREADS, 0, stream>>>(
        x, fw, tw, mg, b0, b1, Wb0, Wb1, h0b, h1b, (float*)d_out, bar);
}
